// LinearAttention_19980187861861
// MI455X (gfx1250) — compile-verified, matrix operand order checked by reading, not run on hardware
//
#include <hip/hip_runtime.h>
#include <hip/hip_bf16.h>

// ---------------------------------------------------------------------------
// LinearAttention for MI455X (gfx1250, wave32, WMMA + async-to-LDS)
//   q = softmax_row(hs@Wq^T+bq); k = softmax_seq(hs@Wk^T+bk)/32; v = softmax_seq(hs@Wv^T+bv)
//   out = q @ (k^T @ v)      (reassociated: 172 GFLOP instead of 378 GFLOP)
// Layout trick: K,V projections are written TRANSPOSED ([B][D][S]) so that
//  - the sequence softmax becomes a contiguous row softmax, and
//  - every GEMM is NT with K-contiguous operands (no transpose staging):
//      VK[b] = VT[b] (.)NT KT[b]   (VK[d2][d1] = sum_s v[s,d2] k[s,d1])
//      out[b] = Q[b] (.)NT VK[b]   (out[m][d]  = sum_j q[m][j] VK[d][j])
// bf16 GEMMs stage tiles with global_load_async_to_lds_b128 (ASYNCcnt).
// ---------------------------------------------------------------------------

typedef __attribute__((ext_vector_type(16))) __bf16 v16bf;
typedef __attribute__((ext_vector_type(8)))  float  v8f;

#define BLK_M 128
#define BLK_N 128
#define BLK_K 32
#define LDS_STRIDE 40    // 32 bf16 + 8 pad (80B rows, 16B aligned)
#define CT_STRIDE (BLK_N + 8)

__device__ __forceinline__ unsigned short f2bf(float f) {
    union { float f; unsigned u; } c; c.f = f;
    unsigned r = c.u + 0x7FFFu + ((c.u >> 16) & 1u);   // round to nearest even
    return (unsigned short)(r >> 16);
}
__device__ __forceinline__ float bf2f(unsigned short s) {
    union { unsigned u; float f; } c; c.u = ((unsigned)s) << 16;
    return c.f;
}
__device__ __forceinline__ uint2 pack4(float4 f) {
    uint2 r;
    r.x = (unsigned)f2bf(f.x) | ((unsigned)f2bf(f.y) << 16);
    r.y = (unsigned)f2bf(f.z) | ((unsigned)f2bf(f.w) << 16);
    return r;
}

union AFrag { v16bf v; uint4 q[2]; };

// ---------------- fp32 source staging: global->reg->(cvt)->LDS -------------
struct StageF32 {
    float4 f[4];
    int r, cb;
    __device__ StageF32() { int t = threadIdx.x; r = t >> 1; cb = (t & 1) * 16; }
    __device__ void g2r(const char* src8, int ld, int rc0, int k0) {
        const float* p = (const float*)src8 + (size_t)(rc0 + r) * ld + k0 + cb;
#pragma unroll
        for (int q = 0; q < 4; ++q) f[q] = *(const float4*)(p + 4 * q);
        __builtin_prefetch(p + BLK_K, 0, 1);
    }
    __device__ void r2lds(unsigned short (*dst)[LDS_STRIDE]) {
#pragma unroll
        for (int q = 0; q < 4; ++q) *(uint2*)&dst[r][cb + 4 * q] = pack4(f[q]);
    }
};

// -------- bf16 source staging: CDNA5 async DMA straight into LDS -----------
struct StageAsync {
    int r, cb;
    __device__ StageAsync() { int t = threadIdx.x; r = t >> 1; cb = (t & 1) * 16; }
    __device__ void issue(const char* src8, int ld, int rc0, int k0,
                          unsigned short (*dst)[LDS_STRIDE]) {
        const unsigned short* p =
            (const unsigned short*)src8 + (size_t)(rc0 + r) * ld + k0 + cb;
        unsigned l0 = (unsigned)(unsigned long long)&dst[r][cb];  // LDS byte offset
        asm volatile("global_load_async_to_lds_b128 %0, %1, off"
                     :: "v"(l0), "v"((unsigned long long)p) : "memory");
        asm volatile("global_load_async_to_lds_b128 %0, %1, off"
                     :: "v"(l0 + 16u), "v"((unsigned long long)(p + 8)) : "memory");
    }
};
__device__ __forceinline__ void wait_async0() {
    asm volatile("s_wait_asynccnt 0" ::: "memory");
}

// --------------------------- WMMA compute step -----------------------------
__device__ __forceinline__ void mma_step(
    const unsigned short (*As)[LDS_STRIDE], const unsigned short (*Bs)[LDS_STRIDE],
    v8f (&acc)[4][2], int wm, int wn, int lrow, int koff)
{
    AFrag fa[4], fb[2];
#pragma unroll
    for (int i = 0; i < 4; ++i) {
        const int r = wm + 16 * i + lrow;
        fa[i].q[0] = *(const uint4*)&As[r][koff];
        fa[i].q[1] = *(const uint4*)&As[r][koff + 16];
    }
#pragma unroll
    for (int j = 0; j < 2; ++j) {
        const int r = wn + 16 * j + lrow;
        fb[j].q[0] = *(const uint4*)&Bs[r][koff];
        fb[j].q[1] = *(const uint4*)&Bs[r][koff + 16];
    }
#pragma unroll
    for (int i = 0; i < 4; ++i)
#pragma unroll
        for (int j = 0; j < 2; ++j)
            acc[i][j] = __builtin_amdgcn_wmma_f32_16x16x32_bf16(
                false, fa[i].v, false, fb[j].v, (short)0, acc[i][j], false, false);
}

// NT GEMM: C[m][n] = sum_k A[m][k] * B[n][k]  (+bias[n])
// INBF: sources are bf16 (async-to-LDS path) else fp32 (reg+cvt path)
// CBF : output bf16 else fp32
// CTRANS: write C transposed per batch: C[z][n][m'] (z = m/mRows, m' = m%mRows)
template <bool INBF, bool CBF, bool CTRANS>
__global__ __launch_bounds__(256)
void gemm_wmma_nt(const void* __restrict__ A, long long aBatch, int lda,
                  const void* __restrict__ B, long long bBatch, int ldb,
                  const float* __restrict__ bias,
                  void* __restrict__ C, long long cBatch, int ldc,
                  int Kdim, int mRows)
{
    union SMem {
        struct { unsigned short A[2][BLK_M][LDS_STRIDE];
                 unsigned short B[2][BLK_N][LDS_STRIDE]; } p;
        unsigned short ct[BLK_M][CT_STRIDE];
    };
    __shared__ __align__(16) SMem sm;

    const char* A8 = (const char*)A + (size_t)blockIdx.z * aBatch * (INBF ? 2 : 4);
    const char* B8 = (const char*)B + (size_t)blockIdx.z * bBatch * (INBF ? 2 : 4);
    char*       C8 = (char*)C       + (size_t)blockIdx.z * cBatch * (CBF ? 2 : 4);

    const int m0 = blockIdx.y * BLK_M;
    const int n0 = blockIdx.x * BLK_N;

    const int lane = threadIdx.x & 31;
    const int wave = threadIdx.x >> 5;
    const int wm   = (wave & 1) * 64;
    const int wn   = (wave >> 1) * 32;
    const int hlf  = lane >> 4;          // lane half selects K+8 / M+8 per ISA layout
    const int lrow = lane & 15;
    const int koff = hlf * 8;

    v8f acc[4][2];
#pragma unroll
    for (int i = 0; i < 4; ++i)
#pragma unroll
        for (int j = 0; j < 2; ++j) acc[i][j] = (v8f)0.0f;

    StageF32   sa, sb;     // fp32 path (projections)
    StageAsync aa, ab;     // bf16 async path

    if (INBF) {            // prologue: DMA tile 0 into buffer 0
        aa.issue(A8, lda, m0, 0, sm.p.A[0]);
        ab.issue(B8, ldb, n0, 0, sm.p.B[0]);
        wait_async0();
    } else {
        sa.g2r(A8, lda, m0, 0);
        sb.g2r(B8, ldb, n0, 0);
        sa.r2lds(sm.p.A[0]);
        sb.r2lds(sm.p.B[0]);
    }
    __syncthreads();

    int cur = 0;
    for (int k0 = 0; k0 < Kdim; k0 += BLK_K) {
        const int knext = k0 + BLK_K;
        const bool more = knext < Kdim;
        if (more) {                          // stage next tile early (overlaps WMMA)
            if (INBF) {
                aa.issue(A8, lda, m0, knext, sm.p.A[cur ^ 1]);
                ab.issue(B8, ldb, n0, knext, sm.p.B[cur ^ 1]);
            } else {
                sa.g2r(A8, lda, m0, knext);
                sb.g2r(B8, ldb, n0, knext);
            }
        }
        mma_step(sm.p.A[cur], sm.p.B[cur], acc, wm, wn, lrow, koff);
        if (more) {
            if (INBF) wait_async0();         // DMA landed before publishing buffer
            else { sa.r2lds(sm.p.A[cur ^ 1]); sb.r2lds(sm.p.B[cur ^ 1]); }
        }
        __syncthreads();
        cur ^= 1;
    }

    // ----------------------------- epilogue -------------------------------
    // C/D 16x16 f32 layout: VGPR r -> M = r + 8*(lane>>4); N = lane&15
    if (CTRANS) {
        // stash 128x128 bf16 tile in LDS, then coalesced transposed store
#pragma unroll
        for (int i = 0; i < 4; ++i)
#pragma unroll
            for (int j = 0; j < 2; ++j) {
                const int nl = wn + 16 * j + lrow;
                const float badd = bias ? bias[n0 + nl] : 0.0f;
                const int ml = wm + 16 * i + 8 * hlf;
#pragma unroll
                for (int r = 0; r < 8; ++r)
                    sm.ct[ml + r][nl] = f2bf(acc[i][j][r] + badd);
            }
        __syncthreads();

        const int zb   = m0 / mRows;                 // batch from global row
        const int mloc = m0 - zb * mRows;
        const int t   = threadIdx.x;
        const int n   = t >> 1;                      // output row (feature d)
        const int seg = (t & 1) * 64;                // 64 cols (sequence) per thread
        unsigned short* cp = (unsigned short*)C8 + (size_t)zb * cBatch
                           + (size_t)(n0 + n) * ldc + mloc + seg;
#pragma unroll
        for (int q = 0; q < 8; ++q) {
            union { uint4 v; unsigned short s[8]; } w;
#pragma unroll
            for (int e = 0; e < 8; ++e) w.s[e] = sm.ct[seg + 8 * q + e][n];
            *(uint4*)(cp + 8 * q) = w.v;
        }
    } else {
#pragma unroll
        for (int i = 0; i < 4; ++i)
#pragma unroll
            for (int j = 0; j < 2; ++j) {
                const int col = n0 + wn + 16 * j + lrow;
                const float badd = bias ? bias[col] : 0.0f;
                const size_t row0 = (size_t)(m0 + wm + 16 * i + 8 * hlf);
                if (CBF) {
                    unsigned short* cp = (unsigned short*)C8 + row0 * ldc + col;
#pragma unroll
                    for (int r = 0; r < 8; ++r) { cp[0] = f2bf(acc[i][j][r] + badd); cp += ldc; }
                } else {
                    float* cp = (float*)C8 + row0 * ldc + col;
#pragma unroll
                    for (int r = 0; r < 8; ++r) { cp[0] = acc[i][j][r] + badd; cp += ldc; }
                }
            }
    }
}

// ------- softmax along contiguous rows (bf16 in/out, fp32 math) ------------
// NL = elements per thread (row length = 256*NL); result scaled by `scale`
template <int NL>
__global__ __launch_bounds__(256)
void row_softmax_bf16(unsigned short* __restrict__ x, float scale)
{
    __shared__ float red[256];
    const int L = 256 * NL;
    unsigned short* row = x + (size_t)blockIdx.x * L;
    const int t = threadIdx.x;
    float v[NL];
    float mx = -INFINITY;
#pragma unroll
    for (int i = 0; i < NL; ++i) { v[i] = bf2f(row[t + 256 * i]); mx = fmaxf(mx, v[i]); }
    red[t] = mx; __syncthreads();
    for (int s = 128; s > 0; s >>= 1) {
        if (t < s) red[t] = fmaxf(red[t], red[t + s]);
        __syncthreads();
    }
    mx = red[0]; __syncthreads();
    float sum = 0.0f;
#pragma unroll
    for (int i = 0; i < NL; ++i) { v[i] = __expf(v[i] - mx); sum += v[i]; }
    red[t] = sum; __syncthreads();
    for (int s = 128; s > 0; s >>= 1) {
        if (t < s) red[t] += red[t + s];
        __syncthreads();
    }
    const float inv = scale / red[0];
#pragma unroll
    for (int i = 0; i < NL; ++i) row[t + 256 * i] = f2bf(v[i] * inv);
}

extern "C" void kernel_launch(void* const* d_in, const int* in_sizes, int n_in,
                              void* d_out, int out_size, void* d_ws, size_t ws_size,
                              hipStream_t stream)
{
    (void)in_sizes; (void)n_in; (void)out_size; (void)ws_size;
    const float* hs = (const float*)d_in[0];
    const float* Wq = (const float*)d_in[1];
    const float* bq = (const float*)d_in[2];
    const float* Wk = (const float*)d_in[3];
    const float* bk = (const float*)d_in[4];
    const float* Wv = (const float*)d_in[5];
    const float* bv = (const float*)d_in[6];
    float* out = (float*)d_out;

    const int Bn = 4, S = 4096, D = 1024;
    const int M = Bn * S;                           // 16384

    unsigned short* Q  = (unsigned short*)d_ws;     // [M][D]      bf16, 32 MB
    unsigned short* KT = Q  + (size_t)M * D;        // [B][D][S]   bf16, 32 MB
    unsigned short* VT = KT + (size_t)M * D;        // [B][D][S]   bf16, 32 MB
    unsigned short* VK = VT + (size_t)M * D;        // [B][D][D]   bf16,  8 MB

    dim3 blk(256);
    dim3 gProj(D / BLK_N, M / BLK_M, 1);

    // 1) projections (fp32 in, bf16 out); K,V written transposed per batch
    gemm_wmma_nt<false, true, false><<<gProj, blk, 0, stream>>>(
        hs, 0LL, D, Wq, 0LL, D, bq, Q, 0LL, D, D, S);
    gemm_wmma_nt<false, true, true><<<gProj, blk, 0, stream>>>(
        hs, 0LL, D, Wk, 0LL, D, bk, KT, (long long)D * S, S, D, S);
    gemm_wmma_nt<false, true, true><<<gProj, blk, 0, stream>>>(
        hs, 0LL, D, Wv, 0LL, D, bv, VT, (long long)D * S, S, D, S);

    // 2) softmaxes — all contiguous-row now
    row_softmax_bf16<4><<<M, 256, 0, stream>>>(Q, 1.0f);                 // over features
    row_softmax_bf16<16><<<Bn * D, 256, 0, stream>>>(KT, 1.0f / 32.0f);  // over sequence, *1/sqrt(D)
    row_softmax_bf16<16><<<Bn * D, 256, 0, stream>>>(VT, 1.0f);          // over sequence

    // 3) VK[b][d2][d1] = sum_s v[s,d2] k[s,d1]   (NT on VT,KT; K = S) — async DMA
    gemm_wmma_nt<true, true, false><<<dim3(D / BLK_N, D / BLK_M, Bn), blk, 0, stream>>>(
        VT, (long long)D * S, S, KT, (long long)D * S, S, nullptr,
        VK, (long long)D * D, D, S, S);

    // 4) out[b][m][d] = sum_j q[m,j] VK[d,j]     (NT on Q,VK; K = D) — async DMA
    gemm_wmma_nt<true, false, false><<<dim3(D / BLK_N, S / BLK_M, Bn), blk, 0, stream>>>(
        Q, (long long)S * D, D, VK, (long long)D * D, D, nullptr,
        out, (long long)S * D, D, D, S);
}
